// PointConv_36180804501844
// MI455X (gfx1250) — matrix operand order repaired and checked by reading, not compiled, not run on hardware
//
#include <hip/hip_runtime.h>
#include <stdint.h>

// ---------------- problem constants (from reference) ----------------
#define B_   16
#define N_   2048
#define K_   32      // neighbors
#define CI_  64      // chin
#define CO_  64      // chout
#define CM_  16      // weightnet out channels
#define HID_ 32      // weightnet hidden
#define TILE_ 16     // query points per workgroup (M tile)

// output tuple layout: [xyz][conv][mask]
#define OUT_XYZ_OFF  0
#define OUT_CONV_OFF (B_ * N_ * 3)                       // 98304
#define OUT_MASK_OFF (OUT_CONV_OFF + B_ * N_ * CO_)      // 2195456

typedef __attribute__((ext_vector_type(16))) __bf16 v16bf;
typedef __attribute__((ext_vector_type(8)))  float  v8f;

__device__ __forceinline__ float swishf(float x) {
    return x / (1.0f + __expf(-x));
}

// --------------------------------------------------------------------
// Kernel 1: brute-force masked KNN (k=32 smallest squared distances).
// One workgroup = 256 query points of one batch; all 2048 candidate
// coordinates staged into LDS via the CDNA5 async global->LDS path
// (GLOBAL_LOAD_ASYNC_TO_LDS_B32, ASYNCcnt), de-interleaved per lane.
// Per-thread top-32 kept in registers via unrolled replace-max scans.
// --------------------------------------------------------------------
__global__ __launch_bounds__(256) void knn_kernel(const float* __restrict__ xyz,
                                                  const unsigned char* __restrict__ mask,
                                                  int* __restrict__ idx_out) {
    const int b = blockIdx.y;
    const int q = blockIdx.x * 256 + threadIdx.x;   // 0..2047

    __shared__ float sx[N_], sy[N_], sz[N_], sp[N_];

    // async copy xyz -> LDS (x/y/z de-interleaved: per-lane LDS scatter)
    const float* gbase = xyz + (size_t)b * N_ * 3;
    for (int p = threadIdx.x; p < N_; p += 256) {
        const unsigned long long ga = (unsigned long long)(uintptr_t)(gbase + p * 3);
        const unsigned ax = (unsigned)(uintptr_t)&sx[p];
        const unsigned ay = (unsigned)(uintptr_t)&sy[p];
        const unsigned az = (unsigned)(uintptr_t)&sz[p];
        asm volatile("global_load_async_to_lds_b32 %0, %1, off"
                     :: "v"(ax), "v"(ga) : "memory");
        asm volatile("global_load_async_to_lds_b32 %0, %1, off offset:4"
                     :: "v"(ay), "v"(ga) : "memory");
        asm volatile("global_load_async_to_lds_b32 %0, %1, off offset:8"
                     :: "v"(az), "v"(ga) : "memory");
        sp[p] = mask[(size_t)b * N_ + p] ? 0.0f : 1e8f;  // masked -> dist 1e8
    }
    asm volatile("s_wait_asynccnt 0x0" ::: "memory");
    __syncthreads();

    const float qx = sx[q], qy = sy[q], qz = sz[q];

    float bd[K_]; int bi[K_];
#pragma unroll
    for (int j = 0; j < K_; ++j) { bd[j] = 3.0e38f; bi[j] = 0; }
    float cmax = 3.0e38f; int cpos = 0;

    for (int c = 0; c < N_; ++c) {
        const float dx = qx - sx[c], dy = qy - sy[c], dz = qz - sz[c];
        const float d = fmaf(dx, dx, fmaf(dy, dy, dz * dz)) + sp[c];
        if (d < cmax) {
            // replace current max (unrolled selects keep bd/bi in VGPRs)
#pragma unroll
            for (int j = 0; j < K_; ++j)
                if (j == cpos) { bd[j] = d; bi[j] = c; }
            cmax = bd[0]; cpos = 0;
#pragma unroll
            for (int j = 1; j < K_; ++j)
                if (bd[j] > cmax) { cmax = bd[j]; cpos = j; }
        }
    }

    int* out = idx_out + ((size_t)b * N_ + q) * K_;
#pragma unroll
    for (int j = 0; j < K_; ++j) out[j] = bi[j];
}

// --------------------------------------------------------------------
// Kernel 2: convert Wl [1024 x 64] f32 -> bf16 (GEMM2 B matrix, L2-hot)
// --------------------------------------------------------------------
__global__ __launch_bounds__(256) void prep_wl_kernel(const float* __restrict__ Wl,
                                                      __bf16* __restrict__ WlBf) {
    const int t = blockIdx.x * 256 + threadIdx.x;
    if (t < CM_ * CI_ * CO_) {
        const int e = t & 15, ln = (t >> 4) & 31, nt = (t >> 9) & 3, s = t >> 11;
        const int k = 8 * (ln >> 4) + (e & 7) + 16 * (e >> 3);
        WlBf[t] = (__bf16)Wl[(s * 32 + k) * CO_ + nt * 16 + (ln & 15)];
    }
}

// --------------------------------------------------------------------
// Kernel 3: fused gather + WeightNet MLP + einsum(WMMA) + linear(WMMA).
// 256 threads (8 waves) per 16-point tile.
//
// WMMA fragment layouts (CDNA5 ISA 7.12.2, wave32):
//  16-bit A [16x32] (MxK): lane = m + 16*((k>>3)&1),
//                          elem = 8*(k>>4) + 2*((k>>1)&3) + (k&1)
//  16-bit B [32x16] (KxN): lane = n + 16*((k>>3)&1), elem = 8*(k>>4) + (k&7)
//  f32 C/D [16x16]:        lanes 0-15: M=r,N=lane; lanes 16-31: M=r+8,N=lane-16
// --------------------------------------------------------------------
__global__ __launch_bounds__(256) void pointconv_kernel(
        const float* __restrict__ xyz, const float* __restrict__ vals,
        const unsigned char* __restrict__ mask, const int* __restrict__ idxbuf,
        const float* __restrict__ W1, const float* __restrict__ b1,
        const float* __restrict__ W2, const float* __restrict__ b2,
        const float* __restrict__ W3, const float* __restrict__ b3,
        const __bf16* __restrict__ WlBf, const float* __restrict__ bl,
        float* __restrict__ out) {
    const int ptile = blockIdx.x;          // 0..127
    const int b     = blockIdx.y;          // 0..15
    const int tid   = threadIdx.x;
    const int wave  = tid >> 5;            // 0..7
    const int lane  = tid & 31;

    // WeightNet params in LDS
    __shared__ float sW1[3][HID_], sB1[HID_];
    __shared__ float sW2[HID_][HID_], sB2[HID_];
    __shared__ float sW3[HID_][CM_], sB3[CM_];
    __shared__ float sBl[CO_];

    // WMMA staging (pre-swizzled fragment layouts)
    __shared__ __align__(32) __bf16 aFrag1[TILE_][4][32][16];  // 64 KB: V^T per point, 4 c-tiles
    __shared__ __align__(32) __bf16 bFrag1[TILE_][32][16];     // 16 KB: W per point (lane=n+16*((k>>3)&1))
    __shared__ __align__(32) __bf16 a2Frag[32][32][16];        // 32 KB: partial[16 x 1024] as A frags
    __shared__ float cScratch[8][16][16];                      //  8 KB: per-wave GEMM2 C tiles

    // ---- load params ----
    for (int t = tid; t < 3 * HID_; t += 256) sW1[t / HID_][t % HID_] = W1[t];
    for (int t = tid; t < HID_ * HID_; t += 256) sW2[t / HID_][t % HID_] = W2[t];
    for (int t = tid; t < HID_ * CM_; t += 256) sW3[t / CM_][t % CM_] = W3[t];
    if (tid < HID_) { sB1[tid] = b1[tid]; sB2[tid] = b2[tid]; }
    if (tid < CM_)  sB3[tid] = b3[tid];
    if (tid < CO_)  sBl[tid] = bl[tid];
    __syncthreads();

    // ---- Phase 1: gather + MLP, write masked fragments ----
    // 512 (point, neighbor) items over 256 threads
    for (int it = tid; it < TILE_ * K_; it += 256) {
        const int pt = it >> 5;            // point in tile
        const int nb = it & (K_ - 1);      // neighbor (== K index of GEMM1)
        const int gi = b * N_ + ptile * TILE_ + pt;
        const int j  = idxbuf[gi * K_ + nb];
        const int gj = b * N_ + j;

        const float dx = xyz[gi * 3 + 0] - xyz[gj * 3 + 0];
        const float dy = xyz[gi * 3 + 1] - xyz[gj * 3 + 1];
        const float dz = xyz[gi * 3 + 2] - xyz[gj * 3 + 2];

        float h1[HID_];
#pragma unroll
        for (int u = 0; u < HID_; ++u)
            h1[u] = swishf(fmaf(sW1[0][u], dx, fmaf(sW1[1][u], dy, fmaf(sW1[2][u], dz, sB1[u]))));
        float h2[HID_];
#pragma unroll
        for (int u = 0; u < HID_; ++u) {
            float s = sB2[u];
#pragma unroll
            for (int t2 = 0; t2 < HID_; ++t2) s = fmaf(h1[t2], sW2[t2][u], s);
            h2[u] = swishf(s);
        }
        const float m = mask[gj] ? 1.0f : 0.0f;

        // W fragment: B-matrix swizzle for k = nb (lane = n + 16*((k>>3)&1))
#pragma unroll
        for (int u = 0; u < CM_; ++u) {
            float s = sB3[u];
#pragma unroll
            for (int t2 = 0; t2 < HID_; ++t2) s = fmaf(h2[t2], sW3[t2][u], s);
            bFrag1[pt][u + 16 * ((nb >> 3) & 1)][8 * (nb >> 4) + (nb & 7)] = (__bf16)(swishf(s) * m);
        }

        // V^T fragment: A-matrix swizzle for k = nb
        const int g = (nb >> 3) & 1;
        const int e = 8 * (nb >> 4) + 2 * ((nb >> 1) & 3) + (nb & 1);
        const float* vrow = vals + (size_t)gj * CI_;
#pragma unroll
        for (int c = 0; c < CI_; ++c)
            aFrag1[pt][c >> 4][(c & 15) + 16 * g][e] = (__bf16)(vrow[c] * m);
    }
    __syncthreads();

    // ---- Phase 2: einsum  partial[pt] = V^T(64x32) * W(32x16), K=32 -> 1 WMMA/tile
    // 8 waves x 8 tiles = 16 pts x 4 c-tiles; pairs of independent tiles are
    // processed together so the D->VALU hazard of one overlaps the other's work.
    // D is scattered directly into GEMM2's A-fragment layout.
    for (int t = 0; t < 8; t += 2) {
        const int pt0 = wave * 2 + ((t + 0) >> 2), ct0 = (t + 0) & 3;
        const int pt1 = wave * 2 + ((t + 1) >> 2), ct1 = (t + 1) & 3;
        const v16bf av0 = *(const v16bf*)&aFrag1[pt0][ct0][lane][0];
        const v16bf bv0 = *(const v16bf*)&bFrag1[pt0][lane][0];
        const v16bf av1 = *(const v16bf*)&aFrag1[pt1][ct1][lane][0];
        const v16bf bv1 = *(const v16bf*)&bFrag1[pt1][lane][0];
        v8f acc0 = {}, acc1 = {};
        acc0 = __builtin_amdgcn_wmma_f32_16x16x32_bf16(false, av0, false, bv0,
                                                       (short)0, acc0, false, false);
        acc1 = __builtin_amdgcn_wmma_f32_16x16x32_bf16(false, av1, false, bv1,
                                                       (short)0, acc1, false, false);
        const int g = lane >> 4, w = lane & 15;
#pragma unroll
        for (int r = 0; r < 8; ++r) {
            const int cl = r + 8 * g;                      // c within tile
            {   // tile 0
                const int kk = (ct0 * 16 + cl) * 16 + w;   // flattened c*16 + w (K of GEMM2)
                const int s  = kk >> 5, kl = kk & 31;
                const int lane2 = pt0 + 16 * ((kl >> 3) & 1);
                const int e2 = 8 * (kl >> 4) + 2 * ((kl >> 1) & 3) + (kl & 1);
                a2Frag[s][lane2][e2] = (__bf16)acc0[r];
            }
            {   // tile 1
                const int kk = (ct1 * 16 + cl) * 16 + w;
                const int s  = kk >> 5, kl = kk & 31;
                const int lane2 = pt1 + 16 * ((kl >> 3) & 1);
                const int e2 = 8 * (kl >> 4) + 2 * ((kl >> 1) & 3) + (kl & 1);
                a2Frag[s][lane2][e2] = (__bf16)acc1[r];
            }
        }
    }
    __syncthreads();

    // ---- Phase 3: conv = partial[16 x 1024] @ Wl[1024 x 64]
    // wave -> (N-tile = wave&3, K-half = wave>>2); 16 WMMAs each, split across
    // two independent accumulators to break the XDL RAW chain.
    {
        const int nt = wave & 3, kh = wave >> 2;
        v8f acc0 = {}, acc1 = {};
        for (int s = kh * 16; s < kh * 16 + 16; s += 2) {
            const v16bf av0 = *(const v16bf*)&a2Frag[s + 0][lane][0];
            const v16bf av1 = *(const v16bf*)&a2Frag[s + 1][lane][0];
            const __bf16* bp0 = WlBf + ((size_t)((s + 0) * 4 + nt) * 32 + lane) * 16;
            const __bf16* bp1 = WlBf + ((size_t)((s + 1) * 4 + nt) * 32 + lane) * 16;
            __builtin_prefetch((const void*)(bp0 + 64 * CO_), 0, 1);  // B tile 2 K-steps ahead
            const v16bf bv0 = *(const v16bf*)bp0;
            const v16bf bv1 = *(const v16bf*)bp1;
            acc0 = __builtin_amdgcn_wmma_f32_16x16x32_bf16(false, av0, false, bv0,
                                                           (short)0, acc0, false, false);
            acc1 = __builtin_amdgcn_wmma_f32_16x16x32_bf16(false, av1, false, bv1,
                                                           (short)0, acc1, false, false);
        }
        const int g = lane >> 4, n = lane & 15;
#pragma unroll
        for (int r = 0; r < 8; ++r) cScratch[wave][r + 8 * g][n] = acc0[r] + acc1[r];
    }
    __syncthreads();

    // ---- combine K-halves, add bias, mask, store ----
    for (int t2 = tid; t2 < TILE_ * CO_; t2 += 256) {
        const int m = t2 >> 6, c = t2 & 63;
        const int nt2 = c >> 4, n = c & 15;
        float v = cScratch[nt2][m][n] + cScratch[nt2 + 4][m][n] + sBl[c];
        const int gi = b * N_ + ptile * TILE_ + m;
        v = mask[gi] ? v : 0.0f;
        out[OUT_CONV_OFF + gi * CO_ + c] = v;
    }
}

// --------------------------------------------------------------------
// Kernel 4: pass-through tuple members (xyz, mask) into output buffer
// --------------------------------------------------------------------
__global__ __launch_bounds__(256) void tail_kernel(const float* __restrict__ xyz,
                                                   const unsigned char* __restrict__ mask,
                                                   float* __restrict__ out) {
    const int t = blockIdx.x * 256 + threadIdx.x;
    if (t < B_ * N_) {
        out[OUT_XYZ_OFF + t * 3 + 0] = xyz[t * 3 + 0];
        out[OUT_XYZ_OFF + t * 3 + 1] = xyz[t * 3 + 1];
        out[OUT_XYZ_OFF + t * 3 + 2] = xyz[t * 3 + 2];
        out[OUT_MASK_OFF + t] = mask[t] ? 1.0f : 0.0f;
    }
}

extern "C" void kernel_launch(void* const* d_in, const int* in_sizes, int n_in,
                              void* d_out, int out_size, void* d_ws, size_t ws_size,
                              hipStream_t stream) {
    const float*         xyz  = (const float*)d_in[0];
    const float*         vals = (const float*)d_in[1];
    const unsigned char* mask = (const unsigned char*)d_in[2];
    const float* W1 = (const float*)d_in[3];
    const float* b1 = (const float*)d_in[4];
    const float* W2 = (const float*)d_in[5];
    const float* b2 = (const float*)d_in[6];
    const float* W3 = (const float*)d_in[7];
    const float* b3 = (const float*)d_in[8];
    const float* Wl = (const float*)d_in[9];
    const float* bl = (const float*)d_in[10];
    float* out = (float*)d_out;

    // workspace: [ idx: B*N*K int (4 MB) ][ WlBf: 1024*64 bf16 (128 KB) ]
    int*    idxbuf = (int*)d_ws;
    __bf16* WlBf   = (__bf16*)((char*)d_ws + (size_t)B_ * N_ * K_ * sizeof(int));

    knn_kernel<<<dim3(N_ / 256, B_), 256, 0, stream>>>(xyz, mask, idxbuf);
    prep_wl_kernel<<<(CM_ * CI_ * CO_ + 255) / 256, 256, 0, stream>>>(Wl, WlBf);
    tail_kernel<<<(B_ * N_ + 255) / 256, 256, 0, stream>>>(xyz, mask, out);
    pointconv_kernel<<<dim3(N_ / TILE_, B_), 256, 0, stream>>>(
        xyz, vals, mask, idxbuf, W1, b1, W2, b2, W3, b3, WlBf, bl, out);
}
